// NodeToHyperedgeLayer_5214090297383
// MI455X (gfx1250) — matrix operand order repaired and checked by reading, not compiled, not run on hardware
//
#include <hip/hip_runtime.h>
#include <hip/hip_bf16.h>

#define IN_DIM   256
#define OUT_DIM  256
#define NUM_HEADS 4
#define HEAD_DIM  64

typedef __attribute__((ext_vector_type(16))) __bf16 v16bf;
typedef __attribute__((ext_vector_type(8)))  __bf16 v8bf;
typedef __attribute__((ext_vector_type(8)))  float  v8f;

// ---------------------------------------------------------------------------
// f32 -> bf16 conversion (A operands for WMMA)
// ---------------------------------------------------------------------------
__global__ void cvt_f32_bf16(const float* __restrict__ x, __bf16* __restrict__ y, int n) {
  int i = blockIdx.x * blockDim.x + threadIdx.x;
  if (i < n) y[i] = (__bf16)x[i];
}

// ---------------------------------------------------------------------------
// Pack W [OUT_DIM, IN_DIM] row-major f32 into WMMA B-fragment order (bf16).
// B = W^T is KxN; ISA 32x16 bf16 B layout: lane l holds N = ntile*16 + (l&15),
// with the 16 bf16 in its 8 VGPRs being K = k0+8*(l>>4)+0..7, k0+16+8*(l>>4)+0..7.
// Bp index = ((ntile*8 + kstep)*32 + lane)*16 + n
// ---------------------------------------------------------------------------
__global__ void pack_w(const float* __restrict__ W, __bf16* __restrict__ Bp) {
  int idx   = blockIdx.x * blockDim.x + threadIdx.x;   // 0 .. 65535
  int n     = idx & 15;
  int lane  = (idx >> 4) & 31;
  int kstep = (idx >> 9) & 7;
  int ntile = idx >> 12;
  int k = kstep * 32 + ((n & 8) << 1) + ((lane >> 4) << 3) + (n & 7);
  Bp[idx] = (__bf16)W[(ntile * 16 + (lane & 15)) * IN_DIM + k];
}

// ---------------------------------------------------------------------------
// Shared helper: build the A fragment for one 16-row M tile / one k-step.
// 16-bit 16x32 A layout: lane holds K = k0+8*half+0..7 and k0+16+8*half+0..7,
// two contiguous 16-byte chunks -> global_load_b128 each.
// ---------------------------------------------------------------------------
__device__ __forceinline__ v16bf load_a_frag(const __bf16* Arow, int k0, int half) {
  v8bf c0 = *(const v8bf*)(Arow + k0 + (half << 3));
  v8bf c1 = *(const v8bf*)(Arow + k0 + 16 + (half << 3));
  v16bf a;
#pragma unroll
  for (int i = 0; i < 8; ++i) { a[i] = c0[i]; a[i + 8] = c1[i]; }
  return a;
}

// ---------------------------------------------------------------------------
// Single-output GEMM: Y = bf16(X) @ W^T + b  (used for Q projection).
// Block = 128 threads (4 waves); wave w owns N tiles 4w..4w+3; 32 WMMA/wave.
// ---------------------------------------------------------------------------
__global__ __launch_bounds__(128)
void gemm_bf16_wmma(const __bf16* __restrict__ A, const __bf16* __restrict__ Bp,
                    const float* __restrict__ bias, float* __restrict__ Y, int M) {
  const int lane  = threadIdx.x & 31;
  const int wave  = threadIdx.x >> 5;
  const int half  = lane >> 4;
  const int l16   = lane & 15;
  const int mBase = blockIdx.x << 4;

  const __bf16* Arow = A + (size_t)(mBase + l16) * IN_DIM;
  const v16bf*  BpV  = (const v16bf*)Bp;

  v8f acc[4] = {};

#pragma unroll
  for (int ks = 0; ks < 8; ++ks) {
    v16bf a = load_a_frag(Arow, ks << 5, half);
#pragma unroll
    for (int nt = 0; nt < 4; ++nt) {
      const int ntile = (wave << 2) + nt;
      v16bf b = BpV[((size_t)ntile * 8 + ks) * 32 + lane];
      acc[nt] = __builtin_amdgcn_wmma_f32_16x16x32_bf16(
          false, a, false, b, (short)0, acc[nt], false, false);
    }
  }

  // C/D layout: VGPR j -> M = 8*half + j ; lane&15 -> N
#pragma unroll
  for (int nt = 0; nt < 4; ++nt) {
    const int ncol = (((wave << 2) + nt) << 4) + l16;
    const float bv = bias[ncol];
    float* yb = Y + (size_t)(mBase + (half << 3)) * OUT_DIM + ncol;
#pragma unroll
    for (int j = 0; j < 8; ++j)
      yb[(size_t)j * OUT_DIM] = acc[nt][j] + bv;
  }
}

// ---------------------------------------------------------------------------
// Fused dual-output GEMM: K = X@Wk^T + bk and V = X@Wv^T + bv from ONE pass
// over A (h_src bf16 read once, not twice). 8 accumulators, 64 WMMA / wave.
// ---------------------------------------------------------------------------
__global__ __launch_bounds__(128)
void gemm_bf16_wmma_dual(const __bf16* __restrict__ A,
                         const __bf16* __restrict__ BpK, const __bf16* __restrict__ BpVw,
                         const float* __restrict__ biasK, const float* __restrict__ biasV,
                         float* __restrict__ YK, float* __restrict__ YV, int M) {
  const int lane  = threadIdx.x & 31;
  const int wave  = threadIdx.x >> 5;
  const int half  = lane >> 4;
  const int l16   = lane & 15;
  const int mBase = blockIdx.x << 4;

  const __bf16* Arow = A + (size_t)(mBase + l16) * IN_DIM;
  const v16bf*  BK   = (const v16bf*)BpK;
  const v16bf*  BV   = (const v16bf*)BpVw;

  v8f accK[4] = {};
  v8f accV[4] = {};

#pragma unroll
  for (int ks = 0; ks < 8; ++ks) {
    v16bf a = load_a_frag(Arow, ks << 5, half);
#pragma unroll
    for (int nt = 0; nt < 4; ++nt) {
      const int ntile = (wave << 2) + nt;
      const size_t fi = ((size_t)ntile * 8 + ks) * 32 + lane;
      v16bf bk = BK[fi];
      v16bf bv = BV[fi];
      accK[nt] = __builtin_amdgcn_wmma_f32_16x16x32_bf16(
          false, a, false, bk, (short)0, accK[nt], false, false);
      accV[nt] = __builtin_amdgcn_wmma_f32_16x16x32_bf16(
          false, a, false, bv, (short)0, accV[nt], false, false);
    }
  }

#pragma unroll
  for (int nt = 0; nt < 4; ++nt) {
    const int ncol = (((wave << 2) + nt) << 4) + l16;
    const float bk = biasK[ncol];
    const float bv = biasV[ncol];
    const size_t rowOff = (size_t)(mBase + (half << 3)) * OUT_DIM + ncol;
    float* yk = YK + rowOff;
    float* yv = YV + rowOff;
#pragma unroll
    for (int j = 0; j < 8; ++j) {
      yk[(size_t)j * OUT_DIM] = accK[nt][j] + bk;
      yv[(size_t)j * OUT_DIM] = accV[nt][j] + bv;
    }
  }
}

// ---------------------------------------------------------------------------
// Edge scores + segment max (ordered-u32 atomicMax).
// Monotonic encoding: f>=0 -> bits|0x80000000 ; f<0 -> ~bits. Init 0 == "empty".
// ---------------------------------------------------------------------------
__global__ void edge_score(const float* __restrict__ Q, const float* __restrict__ K,
                           const int* __restrict__ src, const int* __restrict__ dst,
                           float* __restrict__ score, unsigned* __restrict__ segmax, int E) {
  int t = blockIdx.x * blockDim.x + threadIdx.x;
  if (t >= E * NUM_HEADS) return;
  int e = t >> 2, h = t & 3;
  int s = src[e], d = dst[e];
  const float4* kp = (const float4*)(K + (size_t)s * OUT_DIM + h * HEAD_DIM);
  const float4* qp = (const float4*)(Q + (size_t)d * OUT_DIM + h * HEAD_DIM);
  float acc = 0.f;
#pragma unroll
  for (int i = 0; i < HEAD_DIM / 4; ++i) {
    float4 kv = kp[i], qv = qp[i];
    acc = fmaf(kv.x, qv.x, acc);
    acc = fmaf(kv.y, qv.y, acc);
    acc = fmaf(kv.z, qv.z, acc);
    acc = fmaf(kv.w, qv.w, acc);
  }
  acc *= 0.125f;  // 1/sqrt(HEAD_DIM)
  score[t] = acc;
  unsigned bits = __float_as_uint(acc);
  unsigned ord  = (bits & 0x80000000u) ? ~bits : (bits | 0x80000000u);
  atomicMax(&segmax[d * NUM_HEADS + h], ord);
}

// ---------------------------------------------------------------------------
// e = exp(score - segmax[dst]); overwrite score with e; segment-sum via f32 atomics.
// ---------------------------------------------------------------------------
__global__ void edge_expsum(const int* __restrict__ dst, float* __restrict__ score,
                            const unsigned* __restrict__ segmax, float* __restrict__ segsum,
                            int E) {
  int t = blockIdx.x * blockDim.x + threadIdx.x;
  if (t >= E * NUM_HEADS) return;
  int e = t >> 2, h = t & 3;
  int d = dst[e];
  unsigned u = segmax[d * NUM_HEADS + h];
  float m = 0.f;
  if (u != 0u)
    m = (u & 0x80000000u) ? __uint_as_float(u & 0x7FFFFFFFu) : __uint_as_float(~u);
  float ex = __expf(score[t] - m);
  score[t] = ex;
  atomicAdd(&segsum[d * NUM_HEADS + h], ex);
}

// ---------------------------------------------------------------------------
// out[dst, c] += (e/segsum[dst,h]) * V[src, c]  — one thread per (edge, channel).
// ---------------------------------------------------------------------------
__global__ void edge_aggregate(const int* __restrict__ src, const int* __restrict__ dst,
                               const float* __restrict__ score, const float* __restrict__ segsum,
                               const float* __restrict__ V, float* __restrict__ out) {
  int e = blockIdx.x;
  int c = threadIdx.x;           // 0..255
  int h = c >> 6;                // head
  int s = src[e], d = dst[e];
  float a = score[e * NUM_HEADS + h] / segsum[d * NUM_HEADS + h];
  atomicAdd(&out[(size_t)d * OUT_DIM + c], a * V[(size_t)s * OUT_DIM + c]);
}

// ---------------------------------------------------------------------------
extern "C" void kernel_launch(void* const* d_in, const int* in_sizes, int n_in,
                              void* d_out, int out_size, void* d_ws, size_t ws_size,
                              hipStream_t stream) {
  const float* h_src   = (const float*)d_in[0];
  const float* h_dst   = (const float*)d_in[1];
  const int*   src_idx = (const int*)d_in[2];
  const int*   dst_idx = (const int*)d_in[3];
  const float* Wq = (const float*)d_in[4];
  const float* bq = (const float*)d_in[5];
  const float* Wk = (const float*)d_in[6];
  const float* bk = (const float*)d_in[7];
  const float* Wv = (const float*)d_in[8];
  const float* bv = (const float*)d_in[9];

  const int n_src = in_sizes[0] / IN_DIM;   // 100000
  const int n_dst = in_sizes[1] / IN_DIM;   // 50000
  const int E     = in_sizes[2];            // 300000

  // ---- workspace partition (256B aligned) ----
  char* ws = (char*)d_ws;
  size_t off = 0;
  auto carve = [&](size_t bytes) {
    void* p = ws + off;
    off = (off + bytes + 255) & ~(size_t)255;
    return p;
  };
  __bf16*   Asrc  = (__bf16*)carve((size_t)n_src * IN_DIM * sizeof(__bf16));
  __bf16*   Adst  = (__bf16*)carve((size_t)n_dst * IN_DIM * sizeof(__bf16));
  __bf16*   BpQ   = (__bf16*)carve((size_t)OUT_DIM * IN_DIM * sizeof(__bf16));
  __bf16*   BpK   = (__bf16*)carve((size_t)OUT_DIM * IN_DIM * sizeof(__bf16));
  __bf16*   BpV   = (__bf16*)carve((size_t)OUT_DIM * IN_DIM * sizeof(__bf16));
  float*    Qf    = (float*)carve((size_t)n_dst * OUT_DIM * sizeof(float));
  float*    Kf    = (float*)carve((size_t)n_src * OUT_DIM * sizeof(float));
  float*    Vf    = (float*)carve((size_t)n_src * OUT_DIM * sizeof(float));
  float*    score = (float*)carve((size_t)E * NUM_HEADS * sizeof(float));
  unsigned* smax  = (unsigned*)carve((size_t)n_dst * NUM_HEADS * sizeof(unsigned));
  float*    ssum  = (float*)carve((size_t)n_dst * NUM_HEADS * sizeof(float));

  // ---- init accumulators / output ----
  hipMemsetAsync(d_out, 0, (size_t)n_dst * OUT_DIM * sizeof(float), stream);
  hipMemsetAsync(smax, 0, (size_t)n_dst * NUM_HEADS * sizeof(unsigned), stream);
  hipMemsetAsync(ssum, 0, (size_t)n_dst * NUM_HEADS * sizeof(float), stream);

  // ---- convert inputs to bf16 ----
  {
    int n = n_src * IN_DIM;
    cvt_f32_bf16<<<(n + 255) / 256, 256, 0, stream>>>(h_src, Asrc, n);
    n = n_dst * IN_DIM;
    cvt_f32_bf16<<<(n + 255) / 256, 256, 0, stream>>>(h_dst, Adst, n);
  }

  // ---- pack weights into WMMA B-fragment order ----
  pack_w<<<(OUT_DIM * IN_DIM) / 256, 256, 0, stream>>>(Wq, BpQ);
  pack_w<<<(OUT_DIM * IN_DIM) / 256, 256, 0, stream>>>(Wk, BpK);
  pack_w<<<(OUT_DIM * IN_DIM) / 256, 256, 0, stream>>>(Wv, BpV);

  // ---- WMMA projections: Q alone; K+V fused over one pass of h_src ----
  gemm_bf16_wmma<<<n_dst / 16, 128, 0, stream>>>(Adst, BpQ, bq, Qf, n_dst);
  gemm_bf16_wmma_dual<<<n_src / 16, 128, 0, stream>>>(Asrc, BpK, BpV, bk, bv, Kf, Vf, n_src);

  // ---- edge phase ----
  {
    int nt = E * NUM_HEADS;
    edge_score<<<(nt + 255) / 256, 256, 0, stream>>>(Qf, Kf, src_idx, dst_idx, score, smax, E);
    edge_expsum<<<(nt + 255) / 256, 256, 0, stream>>>(dst_idx, score, smax, ssum, E);
  }
  edge_aggregate<<<E, OUT_DIM, 0, stream>>>(src_idx, dst_idx, score, ssum, Vf, (float*)d_out);
}
